// GCNII_64665027609334
// MI455X (gfx1250) — matrix operand order repaired and checked by reading, not compiled, not run on hardware
//
#include <hip/hip_runtime.h>
#include <math.h>

typedef __attribute__((ext_vector_type(16))) __bf16 v16bf;
typedef __attribute__((ext_vector_type(8)))  float  v8f;
typedef __attribute__((ext_vector_type(4)))  int    v4i;

// pointer-to-int4 in global (AS1) / LDS (AS3) address spaces, as the async builtin expects
typedef __attribute__((address_space(1))) v4i* gas_v4i_ptr;
typedef __attribute__((address_space(3))) v4i* las_v4i_ptr;

#define H_DIM 128
#define KSTEP 32

#if defined(__gfx1250__) && \
    __has_builtin(__builtin_amdgcn_global_load_async_to_lds_b128) && \
    __has_builtin(__builtin_amdgcn_s_wait_asynccnt)
#define USE_ASYNC_LDS 1
#else
#define USE_ASYNC_LDS 0
#endif

// ---------------- degree ----------------
__global__ void init_deg_kernel(float* deg, int N) {
  int i = blockIdx.x * blockDim.x + threadIdx.x;
  if (i < N) deg[i] = 1.0f;            // +1 for self loop
}

__global__ void count_deg_kernel(const int* __restrict__ rows, float* deg, int E) {
  int e = blockIdx.x * blockDim.x + threadIdx.x;
  if (e < E) atomicAdd(&deg[rows[e]], 1.0f);
}

// ------ one-time weight conversion: W[K][128] f32 -> Wt[128][K] bf16 (transposed) ------
__global__ void convert_W_kernel(const float* __restrict__ W, __bf16* __restrict__ Wt, int K) {
  int idx = blockIdx.x * blockDim.x + threadIdx.x;   // over K*128
  if (idx >= K * H_DIM) return;
  int n = idx & (H_DIM - 1);      // consecutive threads -> consecutive n (coalesced read)
  int k = idx >> 7;
  Wt[(size_t)n * K + k] = (__bf16)W[idx];
}

// ---------------- hi = h / deg (self-loop term, also init for scatter) ---------
__global__ void init_hi_kernel(const float* __restrict__ h, const float* __restrict__ deg,
                               float* __restrict__ hi, int N) {
  int idx = blockIdx.x * blockDim.x + threadIdx.x;   // over N * H/4
  int total = N * (H_DIM / 4);
  if (idx >= total) return;
  int node = idx / (H_DIM / 4);
  float inv = 1.0f / deg[node];
  float4 v = ((const float4*)h)[idx];
  v.x *= inv; v.y *= inv; v.z *= inv; v.w *= inv;
  ((float4*)hi)[idx] = v;
}

// ---------------- SpMM scatter: one wave per edge ----------------
__global__ void scatter_kernel(const int* __restrict__ rows, const int* __restrict__ cols,
                               const float* __restrict__ deg,
                               const float* __restrict__ h, float* __restrict__ hi, int E) {
  int gid  = blockIdx.x * blockDim.x + threadIdx.x;
  int e    = gid >> 5;
  int lane = gid & 31;
  if (e >= E) return;
  int r = rows[e];
  int c = cols[e];
  float w = rsqrtf(deg[r]) * rsqrtf(deg[c]);
  float4 v = ((const float4*)(h + (size_t)c * H_DIM))[lane];   // 512B coalesced per edge
  float* dst = hi + (size_t)r * H_DIM + lane * 4;
  atomicAdd(dst + 0, v.x * w);
  atomicAdd(dst + 1, v.y * w);
  atomicAdd(dst + 2, v.z * w);
  atomicAdd(dst + 3, v.w * w);
}

// ---------------- WMMA GEMM ----------------
// MODE 0: out = A[N,K] @ W[K,128] + bias         (fc_in)
// MODE 1: support = (1-alpha)*A + alpha*h0 ;  out = theta*(support@W) + (1-theta)*support ; relu?
// Wbf: bf16 weights transposed, [128][K] row-major (row n = output column, contiguous in k).
template<int MODE>
__global__ void gemm_wmma_kernel(const float* __restrict__ A,
                                 const float* __restrict__ h0,
                                 const __bf16* __restrict__ Wbf,
                                 const float* __restrict__ bias,
                                 float* __restrict__ out,
                                 int Nrows, int K,
                                 float alpha, float theta, int do_relu)
{
  // double-buffered K-chunk of transposed bf16 weights: Wt[buf][n][k]
  // rows are 64B, so the two 16-element halves are 32B aligned for v16bf loads
  __shared__ __align__(64) __bf16 Wt[2][H_DIM][KSTEP];

  const int  tid  = threadIdx.x;          // 0..127 (4 waves)
  const int  wave = tid >> 5;
  const int  lane = tid & 31;
  const int  half = lane >> 4;
  const int  l16  = lane & 15;
  const long rowbase = (long)blockIdx.x * 64 + (long)wave * 16;
  const long m = rowbase + l16;           // row this lane supplies for the A fragment

  // stage one 128x32 bf16 tile (8KB): 128 threads x 4 transfers x 16B
  auto stage = [&](int buf, int k0) {
    for (int i = 0; i < 4; ++i) {
      int flat = (i * 128 + tid) * 16;    // byte offset inside tile
      int n    = flat >> 6;               // 64B per row
      int off  = flat & 63;
      char* g = (char*)Wbf + ((size_t)n * K + k0) * 2 + off;   // const cast away for builtin
      char* l = (char*)&Wt[buf][0][0] + flat;
#if USE_ASYNC_LDS
      __builtin_amdgcn_global_load_async_to_lds_b128(
          (gas_v4i_ptr)(void*)g, (las_v4i_ptr)(void*)l, 0, 0);
#else
      *(uint4*)l = *(const uint4*)g;
#endif
    }
  };

  v8f acc[8];
  for (int t = 0; t < 8; ++t) acc[t] = (v8f){0.f,0.f,0.f,0.f,0.f,0.f,0.f,0.f};

  stage(0, 0);
#if USE_ASYNC_LDS
  __builtin_amdgcn_s_wait_asynccnt(0);
#endif
  __syncthreads();

  int cur = 0;
  for (int k0 = 0; k0 < K; k0 += KSTEP) {
    // kick off DMA of the next weight tile while we compute on the current one
    if (k0 + KSTEP < K) stage(cur ^ 1, k0 + KSTEP);

    // A fragment, contiguous K order (matches B):
    // lanes 0-15: elems 0..15 -> K=k0+0..15
    // lanes16-31: elems 0..15 -> K=k0+16..31
    v16bf afrag;
    const int kA0 = k0 + half * 16;
    const int kA1 = k0 + half * 16 + 8;
    if (m < Nrows) {
      const float* ap = A + (size_t)m * K;
      if (k0 + KSTEP < K) __builtin_prefetch(ap + kA0 + KSTEP, 0, 3);
      if (MODE == 0) {
        for (int j = 0; j < 8; ++j) afrag[j]     = (__bf16)ap[kA0 + j];
        for (int j = 0; j < 8; ++j) afrag[8 + j] = (__bf16)ap[kA1 + j];
      } else {
        const float* hp = h0 + (size_t)m * H_DIM;
        for (int j = 0; j < 8; ++j)
          afrag[j]     = (__bf16)((1.0f - alpha) * ap[kA0 + j] + alpha * hp[kA0 + j]);
        for (int j = 0; j < 8; ++j)
          afrag[8 + j] = (__bf16)((1.0f - alpha) * ap[kA1 + j] + alpha * hp[kA1 + j]);
      }
    } else {
      for (int j = 0; j < 16; ++j) afrag[j] = (__bf16)0.0f;
    }

    // 8 output column tiles; B frag: lane = column (l16), halves split K 0-15 / 16-31
    for (int nt = 0; nt < 8; ++nt) {
      const v16bf bfrag = *(const v16bf*)&Wt[cur][nt * 16 + l16][half * 16];
      acc[nt] = __builtin_amdgcn_wmma_f32_16x16x32_bf16(
          /*neg_a=*/false, afrag, /*neg_b=*/false, bfrag,
          /*c_mod=*/(short)0, acc[nt], /*reuse_a=*/false, /*reuse_b=*/false);
    }

#if USE_ASYNC_LDS
    __builtin_amdgcn_s_wait_asynccnt(0);   // our DMA into the other buffer landed
#endif
    __syncthreads();                       // everyone's DMA landed / reads done
    cur ^= 1;
  }

  // epilogue: D layout -> lane column = l16, VGPR r -> row half*8 + r
  for (int nt = 0; nt < 8; ++nt) {
    const int n = nt * 16 + l16;
    for (int r = 0; r < 8; ++r) {
      long mm = rowbase + half * 8 + r;
      if (mm >= Nrows) continue;
      float v = acc[nt][r];
      if (MODE == 0) {
        v += bias[n];
      } else {
        float s = (1.0f - alpha) * A[(size_t)mm * H_DIM + n]
                + alpha * h0[(size_t)mm * H_DIM + n];
        v = theta * v + (1.0f - theta) * s;
        if (do_relu) v = fmaxf(v, 0.0f);
      }
      out[(size_t)mm * H_DIM + n] = v;
    }
  }
}

// ---------------- launcher ----------------
extern "C" void kernel_launch(void* const* d_in, const int* in_sizes, int n_in,
                              void* d_out, int out_size, void* d_ws, size_t ws_size,
                              hipStream_t stream) {
  (void)n_in; (void)out_size; (void)ws_size;
  const float* x    = (const float*)d_in[0];
  const float* fc_w = (const float*)d_in[1];
  const float* fc_b = (const float*)d_in[2];
  const float* Ws   = (const float*)d_in[3];
  const int*   ei   = (const int*)d_in[4];

  const int F_IN = 256, H = H_DIM;
  const int N = in_sizes[0] / F_IN;
  const int E = in_sizes[4] / 2;
  const int L = in_sizes[3] / (H * H);
  const int* rows = ei;        // edge_index[0] = destinations
  const int* cols = ei + E;    // edge_index[1] = sources

  char*  ws   = (char*)d_ws;
  size_t szNH = (size_t)N * H * sizeof(float);
  float*  deg   = (float*)ws;  ws += (((size_t)N * 4 + 255) / 256) * 256;
  float*  h0    = (float*)ws;  ws += szNH;
  float*  bufA  = (float*)ws;  ws += szNH;
  float*  bufB  = (float*)ws;  ws += szNH;
  __bf16* wfcbf = (__bf16*)ws; ws += (((size_t)F_IN * H * 2 + 255) / 256) * 256;
  __bf16* wlbf  = (__bf16*)ws; // L * H * H bf16

  // degrees (self loop included via init to 1.0)
  init_deg_kernel <<<(N + 255) / 256, 256, 0, stream>>>(deg, N);
  count_deg_kernel<<<(E + 255) / 256, 256, 0, stream>>>(rows, deg, E);

  // one-time transposed bf16 weight conversion
  convert_W_kernel<<<(F_IN * H + 255) / 256, 256, 0, stream>>>(fc_w, wfcbf, F_IN);
  for (int l = 0; l < L; ++l)
    convert_W_kernel<<<(H * H + 255) / 256, 256, 0, stream>>>(
        Ws + (size_t)l * H * H, wlbf + (size_t)l * H * H, H);

  // h0 = x @ fc_w + fc_b
  dim3 gblk((N + 63) / 64);
  gemm_wmma_kernel<0><<<gblk, 128, 0, stream>>>(x, nullptr, wfcbf, fc_b, h0,
                                                N, F_IN, 0.f, 0.f, 0);

  const float ALPHA = 0.1f, LAMDA = 0.5f;
  const float* h = h0;
  float* hi    = bufA;
  float* other = bufB;
  for (int l = 1; l <= L; ++l) {
    float theta = logf(LAMDA / (float)l + 1.0f);
    int spmm_blocks = (int)(((size_t)E * 32 + 255) / 256);
    init_hi_kernel<<<(N * (H / 4) + 255) / 256, 256, 0, stream>>>(h, deg, hi, N);
    scatter_kernel<<<spmm_blocks, 256, 0, stream>>>(rows, cols, deg, h, hi, E);
    float* outp = (l == L) ? (float*)d_out : hi;   // in-place per-row is safe
    gemm_wmma_kernel<1><<<gblk, 128, 0, stream>>>(hi, h0, wlbf + (size_t)(l - 1) * H * H,
                                                  nullptr, outp, N, H,
                                                  ALPHA, theta, (l < L) ? 1 : 0);
    h = hi;                     // new h lives in the hi buffer
    float* t = hi; hi = other; other = t;
  }
}
